// RudyWithMacros_71519795413199
// MI455X (gfx1250) — matrix operand order repaired and checked by reading, not compiled, not run on hardware
//
#include <hip/hip_runtime.h>
#include <hip/hip_bf16.h>

typedef __attribute__((ext_vector_type(16))) _Float16 v16h;
typedef __attribute__((ext_vector_type(8)))  float    v8f;

#define NBX 256
#define NBY 256
#define NBINS (NBX * NBY)
#define BSX 3.90625f
#define BSY 3.90625f
#define XLo 0.0f
#define XHi 1000.0f
#define ROUTING_CAP 152.587890625f   /* 1e7 / 65536 */
#define MACRO_UTIL 10.0f

__device__ __forceinline__ float clampf(float v, float lo, float hi) {
    return fminf(fmaxf(v, lo), hi);
}

// ---------------- Kernel 1: per-net bbox + RUDY weights ----------------
__global__ void net_bbox_kernel(const float* __restrict__ pin_pos,
                                const int* __restrict__ netpin_start,
                                const int* __restrict__ flat_netpin,
                                const float* __restrict__ net_w,
                                float* __restrict__ netp,
                                int n_nets, int num_pins) {
    int n = blockIdx.x * blockDim.x + threadIdx.x;
    if (n >= n_nets) return;
    int s0 = netpin_start[n], s1 = netpin_start[n + 1];
    float xmn = 1e30f, xmx = -1e30f, ymn = 1e30f, ymx = -1e30f;
    for (int p = s0; p < s1; ++p) {
        int pid = flat_netpin[p];
        float x = pin_pos[pid];
        float y = pin_pos[num_pins + pid];
        xmn = fminf(xmn, x); xmx = fmaxf(xmx, x);
        ymn = fminf(ymn, y); ymx = fmaxf(ymx, y);
    }
    xmn = clampf(xmn, XLo, XHi); xmx = clampf(xmx, XLo, XHi);
    ymn = clampf(ymn, XLo, XHi); ymx = clampf(ymx, XLo, XHi);
    float w = xmx - xmn, h = ymx - ymn;
    float wt = net_w[n];
    float wH = (h > 0.0f) ? (wt / h) : 0.0f;   // horizontal demand ~ wt/height
    float wV = (w > 0.0f) ? (wt / w) : 0.0f;
    float* o = netp + (size_t)n * 8;
    o[0] = xmn; o[1] = xmx; o[2] = wH; o[3] = wV;
    o[4] = ymn; o[5] = ymx; o[6] = 0.0f; o[7] = 0.0f;
}

// ---------------- Kernel 2: macro footprints ----------------
__global__ void macro_params_kernel(const float* __restrict__ pos,
                                    const float* __restrict__ nsx,
                                    const float* __restrict__ nsy,
                                    const int* __restrict__ midx,
                                    float* __restrict__ macp,
                                    int n_macros, int num_nodes) {
    int m = blockIdx.x * blockDim.x + threadIdx.x;
    if (m >= n_macros) return;
    int id = midx[m];
    float x = pos[id], y = pos[num_nodes + id];
    float sx = nsx[id], sy = nsy[id];
    float inv_area = 1.0f / (sx * sy);
    float u = MACRO_UTIL * inv_area;           // uH == uV here
    float* o = macp + (size_t)m * 8;
    o[0] = clampf(x, XLo, XHi);       o[1] = clampf(x + sx, XLo, XHi);
    o[2] = u;                         o[3] = u;
    o[4] = clampf(y, XLo, XHi);       o[5] = clampf(y + sy, XLo, XHi);
    o[6] = 0.0f;                      o[7] = 0.0f;
}

// ---------------- Kernel 3: net RUDY as split-K WMMA GEMM ----------------
// h_util[x,y] = sum_n (wH_n*ox[n,x]) * oy[n,y]  ==>  A^T * B, K = nets.
// Block: 128 threads = 4 waves sharing one 64x64 output block. Wave w owns
// M-tile (16 x-bins) x 64 y-bins. The 4 B fragments per chunk are identical
// across waves -> built cooperatively (wave w builds tile w) and exchanged
// via LDS; all 8 ds_load_b128 are issued before the WMMA burst so the loads
// pipeline behind one dscnt wait. Grid: (16 spatial blocks, S K-slices).
__global__ __launch_bounds__(128) void rudy_wmma_kernel(
        const float* __restrict__ netp,
        float* __restrict__ part_h, float* __restrict__ part_v,
        int n_nets, int kslice) {
    __shared__ __align__(16) float sm[32 * 8];        // 32 nets x 8 params
    __shared__ __align__(32) float smB[4 * 32 * 8];   // 4 B fragments (v16h/lane)
    const int tid  = threadIdx.x;
    const int wave = tid >> 5;
    const int lane = tid & 31;
    const int g    = lane >> 4;            // half-wave group
    const int mr   = lane & 15;            // A row within tile / B column
    const int blk  = blockIdx.x;           // 0..15 spatial
    const int xbase = ((blk & 3) * 4 + wave) * 16;
    const int ybase = (blk >> 2) * 64;
    const int slice = blockIdx.y;

    int kbeg = slice * kslice;
    int kend = kbeg + kslice;
    if (kend > n_nets) kend = n_nets;

    v8f acc_h[4], acc_v[4];
#pragma unroll
    for (int t = 0; t < 4; ++t) {
        acc_h[t] = (v8f){0.f,0.f,0.f,0.f,0.f,0.f,0.f,0.f};
        acc_v[t] = (v8f){0.f,0.f,0.f,0.f,0.f,0.f,0.f,0.f};
    }

    const float bxl = (float)(xbase + mr) * BSX;
    const float bxh = bxl + BSX;
    const float byl = (float)(ybase + wave * 16 + mr) * BSY;  // for B build (tile==wave)
    const float byh = byl + BSY;

    for (int c = kbeg; c < kend; c += 32) {
        __syncthreads();                   // previous chunk's LDS reads done
        if (tid < 64) {                    // stage 32 nets (64 float4)
            int net = c + (tid >> 1);
            float4 v = {0.f, 0.f, 0.f, 0.f};
            if (net < kend)
                v = reinterpret_cast<const float4*>(netp)[(size_t)net * 2 + (tid & 1)];
            reinterpret_cast<float4*>(sm)[tid] = v;
        }
        __syncthreads();

        // Build B fragment for tile 'wave' only (shared with other waves).
        // B 32x16, this code's contiguous K order (same as A): element h -> K = 16g+h, column = mr.
        {
            v16h Bv;
#pragma unroll
            for (int h = 0; h < 16; ++h) {
                int k = 16 * g + h;
                const float* p = sm + k * 8;
                float oy = fminf(p[5], byh) - fmaxf(p[4], byl);
                Bv[h] = (_Float16)fmaxf(oy, 0.0f);
            }
            reinterpret_cast<v16h*>(smB)[wave * 32 + lane] = Bv;
        }

        // Build A fragments (A 16x32, this code's contiguous K order (same as B): element h -> K = 16g+h)
        v16h Ah, Av;
#pragma unroll
        for (int h = 0; h < 16; ++h) {
            int k = 16 * g + h;
            const float* p = sm + k * 8;
            float ox = fminf(p[1], bxh) - fmaxf(p[0], bxl);
            ox = fmaxf(ox, 0.0f);
            Ah[h] = (_Float16)(p[2] * ox);
            Av[h] = (_Float16)(p[3] * ox);
        }
        __syncthreads();                   // B fragments visible to all waves

        // Fetch all four B fragments first so the ds_load_b128s pipeline,
        // then issue the 8 WMMAs back-to-back.
        v16h Bfrag[4];
#pragma unroll
        for (int t = 0; t < 4; ++t)
            Bfrag[t] = reinterpret_cast<const v16h*>(smB)[t * 32 + lane];
#pragma unroll
        for (int t = 0; t < 4; ++t) {
            acc_h[t] = __builtin_amdgcn_wmma_f32_16x16x32_f16(
                false, Ah, false, Bfrag[t], (short)0, acc_h[t], false, false);
            acc_v[t] = __builtin_amdgcn_wmma_f32_16x16x32_f16(
                false, Av, false, Bfrag[t], (short)0, acc_v[t], false, false);
        }
    }

    // Write deterministic per-slice partials. C/D layout: VGPR r -> M = r + 8g.
    float* ph = part_h + (size_t)slice * NBINS;
    float* pv = part_v + (size_t)slice * NBINS;
#pragma unroll
    for (int t = 0; t < 4; ++t) {
#pragma unroll
        for (int r = 0; r < 8; ++r) {
            int x = xbase + 8 * g + r;
            int y = ybase + t * 16 + mr;
            ph[x * NBY + y] = acc_h[t][r];
            pv[x * NBY + y] = acc_v[t][r];
        }
    }
}

// ---------------- Kernel 4: reduce partials + macro demand + capacity ----------------
// One block == one x-column of bins -> x-overlap is block-uniform, so macros
// that miss this column are skipped via a uniform (scalar) branch.
__global__ __launch_bounds__(256) void combine_kernel(
        const float* __restrict__ part_h, const float* __restrict__ part_v,
        const float* __restrict__ macp,
        float* __restrict__ h_util, float* __restrict__ v_util,
        int n_macros, int S) {
    __shared__ float sm[1024 * 8];
    __shared__ float sox[1024];
    int tid = threadIdx.x;
    for (int i = tid; i < n_macros * 2; i += blockDim.x)
        reinterpret_cast<float4*>(sm)[i] = reinterpret_cast<const float4*>(macp)[i];
    __syncthreads();

    const int x = blockIdx.x;              // blockDim==NBY -> one column per block
    const int y = tid;
    const int b = x * NBY + y;
    float bxl = x * BSX, bxh = bxl + BSX;
    float byl = y * BSY, byh = byl + BSY;

    // Block-uniform x-overlap per macro.
    for (int m = tid; m < n_macros; m += blockDim.x)
        sox[m] = fmaxf(fminf(sm[m * 8 + 1], bxh) - fmaxf(sm[m * 8 + 0], bxl), 0.0f);
    __syncthreads();

    float hd = 0.0f, vd = 0.0f;
    for (int m = 0; m < n_macros; ++m) {   // fixed order -> deterministic
        float ox = sox[m];
        if (ox > 0.0f) {                   // uniform branch: same for whole block
            const float* p = sm + m * 8;
            float oy = fmaxf(fminf(p[5], byh) - fmaxf(p[4], byl), 0.0f);
            float oo = ox * oy;
            hd += p[2] * oo;
            vd += p[3] * oo;
        }
    }

    float hr = 0.0f, vr = 0.0f;
    for (int s = 0; s < S; ++s) {          // fixed order -> deterministic
        hr += part_h[(size_t)s * NBINS + b];
        vr += part_v[(size_t)s * NBINS + b];
    }
    h_util[b] = hr / (ROUTING_CAP - hd);
    v_util[b] = vr / (ROUTING_CAP - vd);
}

// ---------------- Kernel 5: 3x3 reflect Gaussian blur + max(|h|,|v|) ----------------
__global__ __launch_bounds__(256) void blur_max_kernel(
        const float* __restrict__ h_util, const float* __restrict__ v_util,
        float* __restrict__ out) {
    int b = blockIdx.x * blockDim.x + threadIdx.x;
    int x = b >> 8, y = b & 255;
    float e  = expf(-1.0f / 512.0f);       // sigma = 16, offsets +-1
    float s  = 1.0f / (1.0f + 2.0f * e);
    float wgt[3] = { e * s, s, e * s };
    float hv = 0.0f, vv = 0.0f;
#pragma unroll
    for (int dx = -1; dx <= 1; ++dx) {
        int xx = x + dx;
        xx = (xx < 0) ? -xx : ((xx > 255) ? 510 - xx : xx);   // 'reflect'
        float wx = wgt[dx + 1];
#pragma unroll
        for (int dy = -1; dy <= 1; ++dy) {
            int yy = y + dy;
            yy = (yy < 0) ? -yy : ((yy > 255) ? 510 - yy : yy);
            float wxy = wx * wgt[dy + 1];
            int idx = xx * NBY + yy;
            hv += wxy * h_util[idx];
            vv += wxy * v_util[idx];
        }
    }
    out[b] = fmaxf(fabsf(hv), fabsf(vv));
}

extern "C" void kernel_launch(void* const* d_in, const int* in_sizes, int n_in,
                              void* d_out, int out_size, void* d_ws, size_t ws_size,
                              hipStream_t stream) {
    const float* pos          = (const float*)d_in[0];
    const float* pin_pos      = (const float*)d_in[1];
    const int*   netpin_start = (const int*)d_in[2];
    const int*   flat_netpin  = (const int*)d_in[3];
    const float* net_w        = (const float*)d_in[4];
    const float* nsx          = (const float*)d_in[5];
    const float* nsy          = (const float*)d_in[6];
    const int*   midx         = (const int*)d_in[7];

    const int num_nodes = in_sizes[0] / 2;
    const int num_pins  = in_sizes[1] / 2;
    const int n_nets    = in_sizes[4];
    const int n_macros  = in_sizes[7];

    float* ws = (float*)d_ws;
    size_t off = 0;
    float* netp   = ws + off; off += (size_t)n_nets * 8;
    float* macp   = ws + off; off += (size_t)n_macros * 8;
    float* h_util = ws + off; off += NBINS;
    float* v_util = ws + off; off += NBINS;

    // Split-K factor from remaining workspace (deterministic per process).
    size_t ws_floats = ws_size / sizeof(float);
    long long remain = (ws_floats > off) ? (long long)(ws_floats - off) : 0;
    int S = (int)(remain / (2LL * NBINS));
    if (S > 64) S = 64;
    if (S < 1)  S = 1;                      // requires modest ws; best effort
    float* part_h = ws + off; off += (size_t)S * NBINS;
    float* part_v = ws + off;
    int kslice = (n_nets + S - 1) / S;

    net_bbox_kernel<<<(n_nets + 255) / 256, 256, 0, stream>>>(
        pin_pos, netpin_start, flat_netpin, net_w, netp, n_nets, num_pins);
    macro_params_kernel<<<(n_macros + 255) / 256, 256, 0, stream>>>(
        pos, nsx, nsy, midx, macp, n_macros, num_nodes);
    rudy_wmma_kernel<<<dim3(16, S), 128, 0, stream>>>(
        netp, part_h, part_v, n_nets, kslice);
    combine_kernel<<<NBX, NBY, 0, stream>>>(
        part_h, part_v, macp, h_util, v_util, n_macros, S);
    blur_max_kernel<<<NBINS / 256, 256, 0, stream>>>(
        h_util, v_util, (float*)d_out);
}
